// PrimitiveMixerBlock_73796128080294
// MI455X (gfx1250) — compile-verified, matrix operand order checked by reading, not run on hardware
//
#include <hip/hip_runtime.h>
#include <hip/hip_bf16.h>
#include <cstdint>
#include <cstddef>

// ---------------------------------------------------------------------------
// Types / WMMA helpers (CDNA5 gfx1250, wave32)
// ---------------------------------------------------------------------------
typedef __attribute__((ext_vector_type(16))) __bf16 v16bf;
typedef __attribute__((ext_vector_type(8)))  float  v8f;

struct U32B { uint4 lo, hi; };

// 16-bit A/B fragment: lane L holds row/col (L&15), two contiguous 8-element
// k-runs at kbase and kbase+16 (kbase = ktile*32 + (L>>4)*8). Two 16B loads.
static __device__ __forceinline__ v16bf load_frag16(const __bf16* p) {
  U32B r;
  r.lo = *reinterpret_cast<const uint4*>(p);
  r.hi = *reinterpret_cast<const uint4*>(p + 16);
  return __builtin_bit_cast(v16bf, r);
}

static __device__ __forceinline__ v8f wmma_bf16(v16bf a, v16bf b, v8f c) {
  return __builtin_amdgcn_wmma_f32_16x16x32_bf16(
      /*neg_a=*/false, a, /*neg_b=*/false, b,
      /*c_mod=*/(short)0, c, /*reuse_a=*/false, /*reuse_b=*/false);
}

static __device__ __forceinline__ float sigm(float x) {
  return 1.0f / (1.0f + __expf(-x));
}

// Problem constants
#define DD   512
#define SS   2048
#define BB   8
#define NROW 16384   // B*S

// ---------------------------------------------------------------------------
// Async global->LDS staging (CDNA5 GLOBAL_LOAD_ASYNC_TO_LDS_B128, ASYNCcnt),
// guarded so the file compiles on toolchains without the builtin.
// Builtin signature (probed via diagnostics): (int4_as1*, int4_as3*, Ii, Ii).
// ---------------------------------------------------------------------------
#ifndef __has_builtin
#define __has_builtin(x) 0
#endif
#if __has_builtin(__builtin_amdgcn_global_load_async_to_lds_b128)
#define HAVE_ASYNC_LDS 1
typedef int v4i_t __attribute__((vector_size(16)));
typedef __attribute__((address_space(1))) v4i_t g1_i4;
typedef __attribute__((address_space(3))) v4i_t l3_i4;
#else
#define HAVE_ASYNC_LDS 0
#endif

// Stage `chunks` x 16B from contiguous global memory into LDS, then barrier.
static __device__ __forceinline__ void stage_strip(__bf16* lds,
                                                   const __bf16* gsrc, int tid,
                                                   int nthreads, int chunks) {
#if HAVE_ASYNC_LDS
  for (int c = tid; c < chunks; c += nthreads) {
    __builtin_amdgcn_global_load_async_to_lds_b128(
        (g1_i4*)(gsrc + (size_t)c * 8), (l3_i4*)(lds + (size_t)c * 8), 0, 0);
  }
#if __has_builtin(__builtin_amdgcn_s_wait_asynccnt)
  __builtin_amdgcn_s_wait_asynccnt(0);
#else
  asm volatile("s_wait_asynccnt 0x0" ::: "memory");
#endif
#else
  for (int c = tid; c < chunks; c += nthreads)
    *reinterpret_cast<uint4*>(lds + (size_t)c * 8) =
        *reinterpret_cast<const uint4*>(gsrc + (size_t)c * 8);
#endif
  __syncthreads();
}

// ---------------------------------------------------------------------------
// K0: f32 -> bf16 conversion (grid-stride)
// ---------------------------------------------------------------------------
__global__ void cvt_bf16_kernel(const float* __restrict__ src,
                                __bf16* __restrict__ dst, int n) {
  for (int i = blockIdx.x * blockDim.x + threadIdx.x; i < n;
       i += gridDim.x * blockDim.x)
    dst[i] = (__bf16)src[i];
}

// ---------------------------------------------------------------------------
// K1: fused projection GEMM + activations, LDS-tiled.
// Block = 512 thr (16 waves) computes a 64-row x 256-col output macro-tile:
// A strip (64x512 bf16, 64KB, contiguous) staged once into LDS (async path),
// each wave owns one 16-col tile and reuses every B fragment across 4 row
// tiles (4 accumulators, 4 WMMAs per B load). A fragments are hoisted so the
// scheduler can batch the 8 ds_load_b128 and issue the 4 WMMAs back-to-back.
// Column space = [U(32 tiles) | angles(16) | candidate(32) | out-gate(32)].
// grid = (NROW/64, 7).
// ---------------------------------------------------------------------------
__global__ void __launch_bounds__(512)
proj_kernel(const __bf16* __restrict__ Xb,
            const __bf16* __restrict__ WuB, const __bf16* __restrict__ WaB,
            const __bf16* __restrict__ WcB, const __bf16* __restrict__ WgB,
            const float* __restrict__ bu, const float* __restrict__ ba,
            const float* __restrict__ bc, const float* __restrict__ bg,
            float* __restrict__ U, float* __restrict__ COSA,
            float* __restrict__ SINA, float* __restrict__ CD,
            float* __restrict__ G) {
  __shared__ __align__(16) __bf16 atile[64 * DD];   // 64 KB

  const int tid = threadIdx.x;
  const int r0  = blockIdx.x * 64;
  stage_strip(atile, Xb + (size_t)r0 * DD, tid, 512, 64 * DD / 8);

  const int wave = tid >> 5;
  const int lane = tid & 31;
  const int n    = lane & 15;
  const int half = lane >> 4;
  const int t    = blockIdx.y * 16 + wave;   // 0..111

  const __bf16* W; const float* bias; int e0, seg;
  if (t < 32)      { seg = 0; W = WuB; bias = bu; e0 = t * 16; }
  else if (t < 48) { seg = 1; W = WaB; bias = ba; e0 = (t - 32) * 16; }
  else if (t < 80) { seg = 2; W = WcB; bias = bc; e0 = (t - 48) * 16; }
  else             { seg = 3; W = WgB; bias = bg; e0 = (t - 80) * 16; }

  const __bf16* brow = W + (size_t)(e0 + n) * DD;
  const __bf16* arow = atile + (size_t)n * DD;

  v8f acc[4] = {};
#pragma unroll
  for (int kt = 0; kt < 16; ++kt) {
    const int kb = kt * 32 + half * 8;
    v16bf b = load_frag16(brow + kb);
    v16bf a0 = load_frag16(arow + kb);
    v16bf a1 = load_frag16(arow + (size_t)16 * DD + kb);
    v16bf a2 = load_frag16(arow + (size_t)32 * DD + kb);
    v16bf a3 = load_frag16(arow + (size_t)48 * DD + kb);
    acc[0] = wmma_bf16(a0, b, acc[0]);
    acc[1] = wmma_bf16(a1, b, acc[1]);
    acc[2] = wmma_bf16(a2, b, acc[2]);
    acc[3] = wmma_bf16(a3, b, acc[3]);
  }

  const float bv = bias[e0 + n];
#pragma unroll
  for (int rt = 0; rt < 4; ++rt) {
#pragma unroll
    for (int i = 0; i < 8; ++i) {
      const int m = rt * 16 + half * 8 + i;
      const size_t row = (size_t)(r0 + m);
      const float v = acc[rt][i] + bv;
      if (seg == 0) {
        U[row * DD + e0 + n] = sigm(v);
      } else if (seg == 1) {
        COSA[row * (DD / 2) + e0 + n] = __cosf(v);
        SINA[row * (DD / 2) + e0 + n] = __sinf(v);
      } else if (seg == 2) {
        CD[row * DD + e0 + n] = tanhf(v);
      } else {
        G[row * DD + e0 + n] = sigm(v);
      }
    }
  }
}

// ---------------------------------------------------------------------------
// K2: persistent single-workgroup scan. 1024 threads = 32 waves; wave w owns
// state columns [16w, 16w+16). Wt B-fragments pre-loaded into VGPRs
// (16 x v16bf = 128 VGPRs/lane) so the 2048-step loop touches only LDS + L2
// gate vectors. Two barriers per step (WAR + RAW on the LDS state).
// ---------------------------------------------------------------------------
__global__ void __launch_bounds__(1024)
scan_kernel(const __bf16* __restrict__ WtB, const float* __restrict__ bt,
            const float* __restrict__ U, const float* __restrict__ COSA,
            const float* __restrict__ SINA, const float* __restrict__ CD,
            __bf16* __restrict__ Sb, float* __restrict__ finalState) {
  __shared__ __align__(16) __bf16 st[16 * DD];   // 16 KB, rows 8..15 = zero pad

  const int tid = threadIdx.x;
  for (int i = tid; i < 16 * DD; i += 1024) st[i] = (__bf16)0.0f;
  __syncthreads();

  const int wave = tid >> 5;           // 0..31 -> column tile
  const int lane = tid & 31;
  const int n    = lane & 15;
  const int half = lane >> 4;
  const int e0   = wave * 16;

  // Pre-load this wave's Wt slice as 16 B-fragments (stays in registers).
  v16bf wfrag[16];
  {
    const __bf16* brow = WtB + (size_t)(e0 + n) * DD;
#pragma unroll
    for (int kt = 0; kt < 16; ++kt)
      wfrag[kt] = load_frag16(brow + kt * 32 + half * 8);
  }
  const float btv   = bt[e0 + n];
  const int   jrot  = (e0 + n) >> 1;           // rotation pair index
  const bool  evenc = ((e0 + n) & 1) == 0;

  const __bf16* arow = st + (size_t)n * DD;    // A rows: m = lane & 15

  for (int s = 0; s < SS; ++s) {
    v8f acc = {};
#pragma unroll
    for (int kt = 0; kt < 16; ++kt) {
      v16bf a = load_frag16(arow + kt * 32 + half * 8);   // ds_load_b128 x2
      acc = wmma_bf16(a, wfrag[kt], acc);
    }

    float nv[8];
#pragma unroll
    for (int i = 0; i < 8; ++i) {
      const float tv    = acc[i] + btv;
      const float other = __shfl_xor(tv, 1, 32);   // partner column value
      float res = 0.0f;
      if (half == 0) {                     // real rows b = i (0..7)
        const size_t idx = (size_t)i * SS + s;
        const float c  = COSA[idx * (DD / 2) + jrot];
        const float sn = SINA[idx * (DD / 2) + jrot];
        const float rot = evenc ? (tv * c - other * sn)
                                : (other * sn + tv * c);
        const float g  = U[idx * DD + e0 + n];
        const float cd = CD[idx * DD + e0 + n];
        res = g * rot + (1.0f - g) * cd;
      }
      nv[i] = res;
    }

    __syncthreads();                     // all A reads done (WAR)
    if (half == 0) {
#pragma unroll
      for (int i = 0; i < 8; ++i) {
        const size_t idx = (size_t)i * SS + s;
        const __bf16 hb = (__bf16)nv[i];
        st[(size_t)i * DD + e0 + n] = hb;          // recurrent state (LDS)
        Sb[idx * DD + e0 + n] = hb;                // states for output GEMM
        if (s == SS - 1) finalState[(size_t)i * DD + e0 + n] = nv[i];
      }
    }
    __syncthreads();                     // writes visible (RAW)
  }
}

// ---------------------------------------------------------------------------
// K3: emitted = out_gate * (states @ Wo^T + bo). Same LDS tiling as K1.
// grid = (NROW/64, 2), block = 512 (16 waves -> 32 col tiles, 4 row tiles).
// ---------------------------------------------------------------------------
__global__ void __launch_bounds__(512)
out_kernel(const __bf16* __restrict__ Sb, const __bf16* __restrict__ WoB,
           const float* __restrict__ bo, const float* __restrict__ G,
           float* __restrict__ out) {
  __shared__ __align__(16) __bf16 atile[64 * DD];   // 64 KB

  const int tid = threadIdx.x;
  const int r0  = blockIdx.x * 64;
  stage_strip(atile, Sb + (size_t)r0 * DD, tid, 512, 64 * DD / 8);

  const int wave = tid >> 5;
  const int lane = tid & 31;
  const int n    = lane & 15;
  const int half = lane >> 4;
  const int e0   = (blockIdx.y * 16 + wave) * 16;

  const __bf16* brow = WoB + (size_t)(e0 + n) * DD;
  const __bf16* arow = atile + (size_t)n * DD;

  v8f acc[4] = {};
#pragma unroll
  for (int kt = 0; kt < 16; ++kt) {
    const int kb = kt * 32 + half * 8;
    v16bf b = load_frag16(brow + kb);
    v16bf a0 = load_frag16(arow + kb);
    v16bf a1 = load_frag16(arow + (size_t)16 * DD + kb);
    v16bf a2 = load_frag16(arow + (size_t)32 * DD + kb);
    v16bf a3 = load_frag16(arow + (size_t)48 * DD + kb);
    acc[0] = wmma_bf16(a0, b, acc[0]);
    acc[1] = wmma_bf16(a1, b, acc[1]);
    acc[2] = wmma_bf16(a2, b, acc[2]);
    acc[3] = wmma_bf16(a3, b, acc[3]);
  }

  const float bv = bo[e0 + n];
#pragma unroll
  for (int rt = 0; rt < 4; ++rt) {
#pragma unroll
    for (int i = 0; i < 8; ++i) {
      const int m = rt * 16 + half * 8 + i;
      const size_t row = (size_t)(r0 + m);
      const float v = acc[rt][i] + bv;
      out[row * DD + e0 + n] = G[row * DD + e0 + n] * v;
    }
  }
}

// ---------------------------------------------------------------------------
// Launch
// ---------------------------------------------------------------------------
extern "C" void kernel_launch(void* const* d_in, const int* in_sizes, int n_in,
                              void* d_out, int out_size, void* d_ws,
                              size_t ws_size, hipStream_t stream) {
  const float* x  = (const float*)d_in[0];
  const float* Wu = (const float*)d_in[1];
  const float* bu = (const float*)d_in[2];
  const float* Wt = (const float*)d_in[3];
  const float* bt = (const float*)d_in[4];
  const float* Wa = (const float*)d_in[5];
  const float* ba = (const float*)d_in[6];
  const float* Wc = (const float*)d_in[7];
  const float* bc = (const float*)d_in[8];
  const float* Wg = (const float*)d_in[9];
  const float* bg = (const float*)d_in[10];
  const float* Wo = (const float*)d_in[11];
  const float* bo = (const float*)d_in[12];

  char* ws = (char*)d_ws;
  // Workspace layout (bytes) -- all offsets 16B aligned.
  const size_t XB_OFF  = 0;                         // 16384*512 bf16
  const size_t WU_OFF  = XB_OFF  + (size_t)NROW * DD * 2;
  const size_t WT_OFF  = WU_OFF  + (size_t)DD * DD * 2;
  const size_t WA_OFF  = WT_OFF  + (size_t)DD * DD * 2;
  const size_t WC_OFF  = WA_OFF  + (size_t)(DD / 2) * DD * 2;
  const size_t WG_OFF  = WC_OFF  + (size_t)DD * DD * 2;
  const size_t WO_OFF  = WG_OFF  + (size_t)DD * DD * 2;
  const size_t U_OFF   = WO_OFF  + (size_t)DD * DD * 2;
  const size_t COS_OFF = U_OFF   + (size_t)NROW * DD * 4;
  const size_t SIN_OFF = COS_OFF + (size_t)NROW * (DD / 2) * 4;
  const size_t CD_OFF  = SIN_OFF + (size_t)NROW * (DD / 2) * 4;
  const size_t G_OFF   = CD_OFF  + (size_t)NROW * DD * 4;
  const size_t SB_OFF  = G_OFF   + (size_t)NROW * DD * 4;

  __bf16* Xb  = (__bf16*)(ws + XB_OFF);
  __bf16* WuB = (__bf16*)(ws + WU_OFF);
  __bf16* WtB = (__bf16*)(ws + WT_OFF);
  __bf16* WaB = (__bf16*)(ws + WA_OFF);
  __bf16* WcB = (__bf16*)(ws + WC_OFF);
  __bf16* WgB = (__bf16*)(ws + WG_OFF);
  __bf16* WoB = (__bf16*)(ws + WO_OFF);
  float*  U    = (float*)(ws + U_OFF);
  float*  COSA = (float*)(ws + COS_OFF);
  float*  SINA = (float*)(ws + SIN_OFF);
  float*  CD   = (float*)(ws + CD_OFF);
  float*  G    = (float*)(ws + G_OFF);
  __bf16* Sb   = (__bf16*)(ws + SB_OFF);

  float* emitted    = (float*)d_out;
  float* finalState = (float*)d_out + (size_t)NROW * DD;   // tail: (8,512)

  auto cvt = [&](const float* s, __bf16* d, int n) {
    int blocks = (n + 255) / 256;
    if (blocks > 32768) blocks = 32768;
    cvt_bf16_kernel<<<blocks, 256, 0, stream>>>(s, d, n);
  };
  cvt(x,  Xb,  NROW * DD);
  cvt(Wu, WuB, DD * DD);
  cvt(Wt, WtB, DD * DD);
  cvt(Wa, WaB, (DD / 2) * DD);
  cvt(Wc, WcB, DD * DD);
  cvt(Wg, WgB, DD * DD);
  cvt(Wo, WoB, DD * DD);

  proj_kernel<<<dim3(NROW / 64, 7), 512, 0, stream>>>(
      Xb, WuB, WaB, WcB, WgB, bu, ba, bc, bg, U, COSA, SINA, CD, G);

  scan_kernel<<<1, 1024, 0, stream>>>(WtB, bt, U, COSA, SINA, CD, Sb,
                                      finalState);

  out_kernel<<<dim3(NROW / 64, 2), 512, 0, stream>>>(Sb, WoB, bo, G, emitted);
}
